// CustomHeadMultiHeadAttention_81664508166673
// MI455X (gfx1250) — matrix operand order repaired and checked by reading, not compiled, not run on hardware
//
#include <hip/hip_runtime.h>
#include <hip/hip_bf16.h>

// ---------------------------------------------------------------------------
// MHA block for MI455X (gfx1250, wave32, WMMA, async-to-LDS).
// cast->bf16 (weights pre-transposed), 3x double-buffered WMMA GEMM (QKV),
// flash attention (WMMA QK^T + PV, online softmax), erf-GELU + 2-class head.
// ---------------------------------------------------------------------------

typedef __attribute__((ext_vector_type(16))) __bf16 v16bf;
typedef __attribute__((ext_vector_type(8)))  float  v8f;
typedef int v4i_vec __attribute__((vector_size(4 * sizeof(int))));

#define HIDDEN 1024
#define SEQ    2048
#define BATCH  8
#define NH     8
#define DK     128
#define ROWS   (BATCH * SEQ)   // 16384

// 16-byte global -> LDS copy. Prefer the CDNA5 async-DMA path (ASYNCcnt);
// fall back to a synchronous VGPR round-trip if the builtin is absent.
__device__ __forceinline__ void async_cp16(const unsigned short* g,
                                           unsigned short* l) {
#if __has_builtin(__builtin_amdgcn_global_load_async_to_lds_b128)
  __builtin_amdgcn_global_load_async_to_lds_b128(
      (v4i_vec*)(void*)g, (v4i_vec*)(void*)l, 0, 0);
#else
  *(uint4*)l = *(const uint4*)g;
#endif
}

__device__ __forceinline__ void async_wait0() {
#if __has_builtin(__builtin_amdgcn_s_wait_asynccnt)
  __builtin_amdgcn_s_wait_asynccnt(0);
#elif __has_builtin(__builtin_amdgcn_global_load_async_to_lds_b128)
  asm volatile("s_wait_asynccnt 0x0" ::: "memory");
#endif
}

struct Frag32B { uint4 lo, hi; };

// Load a 16x32 bf16 A/B-style fragment from LDS. `p` is the per-lane pointer
// (row start + 8*half elements). lo = K[8h..8h+7], hi = K[16+8h..16+8h+7].
__device__ __forceinline__ v16bf ld_frag_lds(const unsigned short* p) {
  Frag32B f;
  f.lo = *(const uint4*)(p);
  f.hi = *(const uint4*)(p + 16);
  return __builtin_bit_cast(v16bf, f);
}

__device__ __forceinline__ unsigned short f2bf(float x) {
  unsigned int u = __builtin_bit_cast(unsigned int, x);
  unsigned int r = u + 0x7FFFu + ((u >> 16) & 1u);   // round-to-nearest-even
  return (unsigned short)(r >> 16);
}

// ---------------------------------------------------------------------------
// elementwise f32 -> bf16 cast (activations)
// ---------------------------------------------------------------------------
__global__ void cast_bf16_kernel(const float* __restrict__ in,
                                 unsigned short* __restrict__ out, int n) {
  int i = blockIdx.x * blockDim.x + threadIdx.x;
  int stride = gridDim.x * blockDim.x;
  for (; i < n; i += stride) out[i] = f2bf(in[i]);
}

// f32 [k][n] -> bf16 transposed [n][k] (one-time; moves the B transpose out
// of the GEMM hot loop so B tiles stage with contiguous b128 async copies)
__global__ void cast_transpose_kernel(const float* __restrict__ W,
                                      unsigned short* __restrict__ Wt) {
  int idx = blockIdx.x * blockDim.x + threadIdx.x;   // coalesced read
  int k = idx >> 10, n = idx & 1023;
  Wt[(size_t)n * HIDDEN + k] = f2bf(W[idx]);
}

// ---------------------------------------------------------------------------
// QKV GEMM: [16384 x 1024] (bf16) x Wt[1024 n][1024 k] (bf16) + bias.
// Tile BM=128, BN=128, BK=32, double-buffered LDS, async-to-LDS staging.
// 8 waves, each owns 32x64 (2x4 WMMA tiles).
// TSTORE==0 -> Out[(b*8+h)*2048+s][d]   (Q, K)
// TSTORE==1 -> Out[(b*8+h)*128+d][s]    (V, pre-transposed for PV)
// ---------------------------------------------------------------------------
#define BM 128
#define BN 128
#define BK 32
#define ASTR 48   // bf16 elems per LDS row (32 + 16 pad, 96B rows, 16B aligned)
#define BSTR 48
#define NKI (HIDDEN / BK)

template <int TSTORE>
__global__ __launch_bounds__(256) void qkv_gemm_kernel(
    const unsigned short* __restrict__ X,
    const unsigned short* __restrict__ Wt,
    const float* __restrict__ bias, unsigned short* __restrict__ Out) {
  __shared__ unsigned short As[2][BM * ASTR];
  __shared__ unsigned short Bs[2][BN * BSTR];

  const int tid  = threadIdx.x;
  const int n0   = blockIdx.x * BN;
  const int m0   = blockIdx.y * BM;
  const int w    = tid >> 5;
  const int lane = tid & 31;
  const int hl   = lane >> 4;
  const int r    = lane & 15;
  const int wm   = w >> 1;        // 4 waves along M
  const int wn   = w & 1;         // 2 waves along N

  // staging role: 2 threads per row, 16 elems (32B) each
  const int srow = tid >> 1;
  const int sh16 = (tid & 1) * 16;
  const unsigned short* gA = X  + (size_t)(m0 + srow) * HIDDEN + sh16;
  const unsigned short* gB = Wt + (size_t)(n0 + srow) * HIDDEN + sh16;
  unsigned short* lA = &As[0][0] + srow * ASTR + sh16;
  unsigned short* lB = &Bs[0][0] + srow * BSTR + sh16;

  v8f acc[2][4];
#pragma unroll
  for (int mt = 0; mt < 2; ++mt)
#pragma unroll
    for (int nt = 0; nt < 4; ++nt) acc[mt][nt] = {};

  // prologue: stage tile 0 into buffer 0
  async_cp16(gA, lA);
  async_cp16(gA + 8, lA + 8);
  async_cp16(gB, lB);
  async_cp16(gB + 8, lB + 8);
  async_wait0();
  __syncthreads();

  for (int i = 0; i < NKI; ++i) {
    const int cur = i & 1;
    if (i + 1 < NKI) {   // issue async staging of next tile into other buffer
      const int k1 = (i + 1) * BK;
      const int nxt = cur ^ 1;
      async_cp16(gA + k1,     lA + nxt * BM * ASTR);
      async_cp16(gA + k1 + 8, lA + nxt * BM * ASTR + 8);
      async_cp16(gB + k1,     lB + nxt * BN * BSTR);
      async_cp16(gB + k1 + 8, lB + nxt * BN * BSTR + 8);
    }

    v16bf a[2], b[4];
#pragma unroll
    for (int mt = 0; mt < 2; ++mt)
      a[mt] = ld_frag_lds(&As[cur][0] + (wm * 32 + mt * 16 + r) * ASTR + 8 * hl);
#pragma unroll
    for (int nt = 0; nt < 4; ++nt)
      b[nt] = ld_frag_lds(&Bs[cur][0] + (wn * 64 + nt * 16 + r) * BSTR + 8 * hl);
#pragma unroll
    for (int mt = 0; mt < 2; ++mt)
#pragma unroll
      for (int nt = 0; nt < 4; ++nt)
        acc[mt][nt] = __builtin_amdgcn_wmma_f32_16x16x32_bf16(
            false, a[mt], false, b[nt], (short)0, acc[mt][nt], false, false);

    async_wait0();
    __syncthreads();
  }

  // epilogue: bias add, cast to bf16, scatter into head-major layout
#pragma unroll
  for (int mt = 0; mt < 2; ++mt) {
#pragma unroll
    for (int nt = 0; nt < 4; ++nt) {
      int col = n0 + wn * 64 + nt * 16 + r;
      float bv = bias[col];
      int h = col >> 7, d = col & 127;
#pragma unroll
      for (int j = 0; j < 8; ++j) {
        int row = m0 + wm * 32 + mt * 16 + j + 8 * hl;
        int bb = row >> 11, s = row & 2047;
        float v = acc[mt][nt][j] + bv;
        size_t addr;
        if (TSTORE) addr = (size_t)((bb * NH + h) * DK + d) * SEQ + s;
        else        addr = (size_t)((bb * NH + h) * SEQ + s) * DK + d;
        Out[addr] = f2bf(v);
      }
    }
  }
}

// ---------------------------------------------------------------------------
// Flash attention. grid = (16 q-tiles, 64 batch*head). 8 waves/WG; wave owns
// 16 query rows (Q frags pinned in VGPRs). Streams 64-key K/V tiles via
// async-to-LDS; prefetches the next tile while computing.
// ---------------------------------------------------------------------------
#define KTILE 64
#define KSTR  136   // 128 + 8 pad (272B rows, 16B aligned)
#define VSTR  72    // 64 + 8 pad  (144B rows)
#define PSTR  72
#define NKT   (SEQ / KTILE)

__global__ __launch_bounds__(256) void attn_kernel(
    const unsigned short* __restrict__ Q, const unsigned short* __restrict__ K,
    const unsigned short* __restrict__ V, float* __restrict__ merged) {
  __shared__ unsigned short Ks[KTILE * KSTR];   // [key][d]
  __shared__ unsigned short Vt[DK * VSTR];      // [d][key]
  __shared__ unsigned short Ps[8 * 16 * PSTR];  // per-wave P tile [16][64]

  const int tid  = threadIdx.x;
  const int w    = tid >> 5;
  const int lane = tid & 31;
  const int hl   = lane >> 4;
  const int r    = lane & 15;
  const int qt   = blockIdx.x;
  const int bh   = blockIdx.y;
  const int qbase = qt * 128 + w * 16;
  const float scale = 0.08838834764831845f;   // 1/sqrt(128)

  // staging roles
  const int kkey = tid >> 2, kseg = tid & 3;   // K: 32 elems/thread
  const int vd   = tid >> 1, vseg = tid & 1;   // V: 32 elems/thread
  const unsigned short* gK =
      K + ((size_t)bh * SEQ + kkey) * DK + 32 * kseg;
  const unsigned short* gV =
      V + ((size_t)bh * DK + vd) * SEQ + 32 * vseg;
  unsigned short* lK = Ks + kkey * KSTR + 32 * kseg;
  unsigned short* lV = Vt + vd * VSTR + 32 * vseg;

  // Q fragments for this wave's 16 rows: 4 K-chunks of 32 (d = 0..127)
  uint4 qlo[4], qhi[4];
#pragma unroll
  for (int c = 0; c < 4; ++c) {
    const uint4* p = (const uint4*)(Q + ((size_t)bh * SEQ + qbase + r) * DK +
                                    32 * c + 8 * hl);
    qlo[c] = p[0];
    qhi[c] = p[2];
  }

  float mrow[8], lrow[8];
  v8f O[8];
#pragma unroll
  for (int j = 0; j < 8; ++j) { mrow[j] = -1e30f; lrow[j] = 0.0f; }
#pragma unroll
  for (int nt = 0; nt < 8; ++nt) O[nt] = {};

  for (int kt = 0; kt < NKT; ++kt) {
    __syncthreads();   // previous tile fully consumed before restage
    {  // async stage K tile [64 keys][128 d]
      const unsigned short* g = gK + (size_t)kt * KTILE * DK;
#pragma unroll
      for (int i = 0; i < 4; ++i) async_cp16(g + 8 * i, lK + 8 * i);
    }
    {  // async stage V tile [128 d][64 keys] (V already stored [d][s])
      const unsigned short* g = gV + kt * KTILE;
#pragma unroll
      for (int i = 0; i < 4; ++i) async_cp16(g + 8 * i, lV + 8 * i);
    }
    async_wait0();
    __syncthreads();

    if (kt + 1 < NKT) {  // warm L2 for the next tile while we compute
      __builtin_prefetch(gK + (size_t)(kt + 1) * KTILE * DK, 0, 1);
      __builtin_prefetch(gV + (kt + 1) * KTILE, 0, 1);
    }

    // S = Q K^T : 4 key-tiles of 16, K-dim = 128 in 4 chunks of 32
    v8f Sacc[4];
#pragma unroll
    for (int nt = 0; nt < 4; ++nt) Sacc[nt] = {};
#pragma unroll
    for (int c = 0; c < 4; ++c) {
      Frag32B fa = { qlo[c], qhi[c] };
      v16bf a = __builtin_bit_cast(v16bf, fa);
#pragma unroll
      for (int nt = 0; nt < 4; ++nt) {
        v16bf b = ld_frag_lds(Ks + (nt * 16 + r) * KSTR + 32 * c + 8 * hl);
        Sacc[nt] = __builtin_amdgcn_wmma_f32_16x16x32_bf16(
            false, a, false, b, (short)0, Sacc[nt], false, false);
      }
    }

    // online softmax: row j(+8*hl) lives across the lanes of the 16-group
    float tmax[8];
#pragma unroll
    for (int j = 0; j < 8; ++j) {
      float v = Sacc[0][j] * scale;
#pragma unroll
      for (int nt = 1; nt < 4; ++nt) v = fmaxf(v, Sacc[nt][j] * scale);
      tmax[j] = v;
    }
#pragma unroll
    for (int off = 1; off < 16; off <<= 1)
#pragma unroll
      for (int j = 0; j < 8; ++j)
        tmax[j] = fmaxf(tmax[j], __shfl_xor(tmax[j], off, 32));

    float cj[8], rsum[8];
#pragma unroll
    for (int j = 0; j < 8; ++j) {
      float mnew = fmaxf(mrow[j], tmax[j]);
      cj[j] = __expf(mrow[j] - mnew);
      mrow[j] = mnew;
      rsum[j] = 0.0f;
    }
#pragma unroll
    for (int nt = 0; nt < 4; ++nt)
#pragma unroll
      for (int j = 0; j < 8; ++j) {
        float p = __expf(Sacc[nt][j] * scale - mrow[j]);
        Sacc[nt][j] = p;
        rsum[j] += p;
      }
#pragma unroll
    for (int off = 1; off < 16; off <<= 1)
#pragma unroll
      for (int j = 0; j < 8; ++j) rsum[j] += __shfl_xor(rsum[j], off, 32);
#pragma unroll
    for (int j = 0; j < 8; ++j) lrow[j] = lrow[j] * cj[j] + rsum[j];

    // P -> LDS (per-wave, transposes column frags into A-fragment rows)
    unsigned short* Pw = Ps + w * 16 * PSTR;
#pragma unroll
    for (int nt = 0; nt < 4; ++nt)
#pragma unroll
      for (int j = 0; j < 8; ++j)
        Pw[(j + 8 * hl) * PSTR + nt * 16 + r] = f2bf(Sacc[nt][j]);

    // rescale O by exp(m_old - m_new)
#pragma unroll
    for (int nt = 0; nt < 8; ++nt)
#pragma unroll
      for (int j = 0; j < 8; ++j) O[nt][j] *= cj[j];

    // O += P (16x64) @ V (64x128): K-dim 64 in 2 chunks of 32
#pragma unroll
    for (int c2 = 0; c2 < 2; ++c2) {
      v16bf a = ld_frag_lds(Pw + r * PSTR + 32 * c2 + 8 * hl);
#pragma unroll
      for (int nt = 0; nt < 8; ++nt) {
        v16bf b = ld_frag_lds(Vt + (nt * 16 + r) * VSTR + 32 * c2 + 8 * hl);
        O[nt] = __builtin_amdgcn_wmma_f32_16x16x32_bf16(
            false, a, false, b, (short)0, O[nt], false, false);
      }
    }
  }

  // normalize and store merged heads: merged[b][s][h*128+d], fp32
  const int b = bh >> 3, h = bh & 7;
#pragma unroll
  for (int j = 0; j < 8; ++j) {
    float inv = 1.0f / lrow[j];
    int s = qbase + j + 8 * hl;
#pragma unroll
    for (int nt = 0; nt < 8; ++nt) {
      int d = nt * 16 + r;
      merged[((size_t)(b * SEQ + s)) * HIDDEN + h * DK + d] = O[nt][j] * inv;
    }
  }
}

// ---------------------------------------------------------------------------
// classifier: exact-erf GELU then [1024] x [1024,2] + bias. One wave per row,
// float4-vectorized.
// ---------------------------------------------------------------------------
__global__ __launch_bounds__(256) void cls_kernel(
    const float* __restrict__ merged, const float* __restrict__ Wc,
    const float* __restrict__ bc, float* __restrict__ out) {
  const int w = threadIdx.x >> 5, lane = threadIdx.x & 31;
  const size_t row = (size_t)blockIdx.x * 8 + w;
  const float4* mr4 = (const float4*)(merged + row * HIDDEN);
  const float4* wc4 = (const float4*)Wc;
  float a0 = 0.0f, a1 = 0.0f;
#pragma unroll
  for (int it = 0; it < 8; ++it) {
    int i4 = lane + it * 32;            // float4 index within the row
    float4 m  = mr4[i4];
    float4 w0 = wc4[i4 * 2 + 0];        // rows 4*i4, 4*i4+1 (c0,c1 pairs)
    float4 w1 = wc4[i4 * 2 + 1];        // rows 4*i4+2, 4*i4+3
    float gx = 0.5f * m.x * (1.0f + erff(m.x * 0.7071067811865475f));
    float gy = 0.5f * m.y * (1.0f + erff(m.y * 0.7071067811865475f));
    float gz = 0.5f * m.z * (1.0f + erff(m.z * 0.7071067811865475f));
    float gw = 0.5f * m.w * (1.0f + erff(m.w * 0.7071067811865475f));
    a0 += gx * w0.x + gy * w0.z + gz * w1.x + gw * w1.z;
    a1 += gx * w0.y + gy * w0.w + gz * w1.y + gw * w1.w;
  }
#pragma unroll
  for (int off = 16; off > 0; off >>= 1) {
    a0 += __shfl_xor(a0, off, 32);
    a1 += __shfl_xor(a1, off, 32);
  }
  if (lane == 0) {
    out[row * 2 + 0] = a0 + bc[0];
    out[row * 2 + 1] = a1 + bc[1];
  }
}

// ---------------------------------------------------------------------------
extern "C" void kernel_launch(void* const* d_in, const int* in_sizes, int n_in,
                              void* d_out, int out_size, void* d_ws,
                              size_t ws_size, hipStream_t stream) {
  (void)in_sizes; (void)n_in; (void)out_size; (void)ws_size;
  const float* x  = (const float*)d_in[0];
  const float* Wq = (const float*)d_in[1];
  const float* bq = (const float*)d_in[2];
  const float* Wk = (const float*)d_in[3];
  const float* bk = (const float*)d_in[4];
  const float* Wv = (const float*)d_in[5];
  const float* bv = (const float*)d_in[6];
  const float* Wc = (const float*)d_in[7];
  const float* bc = (const float*)d_in[8];
  float* out = (float*)d_out;

  char* ws = (char*)d_ws;
  unsigned short* Xbf = (unsigned short*)(ws + 0);           //  32 MB
  unsigned short* Wqb = (unsigned short*)(ws + 33554432);    //   2 MB (transposed)
  unsigned short* Wkb = (unsigned short*)(ws + 35651584);    //   2 MB (transposed)
  unsigned short* Wvb = (unsigned short*)(ws + 37748736);    //   2 MB (transposed)
  unsigned short* Qb  = (unsigned short*)(ws + 39845888);    //  32 MB
  unsigned short* Kb  = (unsigned short*)(ws + 73400320);    //  32 MB
  unsigned short* Vb  = (unsigned short*)(ws + 106954752);   //  32 MB
  float*          merged = (float*)(ws + 140509184);         //  64 MB

  cast_bf16_kernel<<<4096, 256, 0, stream>>>(x, Xbf, ROWS * HIDDEN);
  cast_transpose_kernel<<<4096, 256, 0, stream>>>(Wq, Wqb);
  cast_transpose_kernel<<<4096, 256, 0, stream>>>(Wk, Wkb);
  cast_transpose_kernel<<<4096, 256, 0, stream>>>(Wv, Wvb);

  dim3 gg(HIDDEN / BN, ROWS / BM);   // (8, 128)
  qkv_gemm_kernel<0><<<gg, 256, 0, stream>>>(Xbf, Wqb, bq, Qb);
  qkv_gemm_kernel<0><<<gg, 256, 0, stream>>>(Xbf, Wkb, bk, Kb);
  qkv_gemm_kernel<1><<<gg, 256, 0, stream>>>(Xbf, Wvb, bv, Vb);

  dim3 ga(SEQ / 128, BATCH * NH);    // (16, 64)
  attn_kernel<<<ga, 256, 0, stream>>>(Qb, Kb, Vb, merged);

  cls_kernel<<<ROWS / 8, 256, 0, stream>>>(merged, Wc, bc, out);
}
